// ResonanceGatedAttention_31877247270954
// MI455X (gfx1250) — matrix operand order repaired and checked by reading, not compiled, not run on hardware
//
#include <hip/hip_runtime.h>
#include <hip/hip_bf16.h>
#include <math.h>

// ---------------- types ----------------
typedef __attribute__((ext_vector_type(16))) _Float16 v16h;
typedef __attribute__((ext_vector_type(8)))  float    v8f;
typedef __attribute__((ext_vector_type(4)))  unsigned int u32x4;
typedef __attribute__((ext_vector_type(4)))  float    f32x4;
typedef __attribute__((ext_vector_type(4)))  _Float16 v4h;

#define B_  4
#define S_  1024
#define D_  1024
#define H_  16
#define DH_ 64

union Frag16 { v16h h; u32x4 u[2]; };

__device__ __forceinline__ v8f wmma_f16(const Frag16& a, const Frag16& b, v8f c) {
  // v_wmma_f32_16x16x32_f16  (8-arg form: neg_a, A, neg_b, B, c_mod, C, reuse_a, reuse_b)
  return __builtin_amdgcn_wmma_f32_16x16x32_f16(false, a.h, false, b.h, (short)0, c, false, false);
}

// ---------------- 1) f32 -> f16 convert (vectorized x4) ----------------
__global__ void cvt_f32_f16(const f32x4* __restrict__ src, v4h* __restrict__ dst, int n4) {
  int i = blockIdx.x * blockDim.x + threadIdx.x;
  if (i < n4) {
    f32x4 v = src[i];
    v4h h;
    h.x = (_Float16)v.x; h.y = (_Float16)v.y; h.z = (_Float16)v.z; h.w = (_Float16)v.w;
    dst[i] = h;
  }
}

// ---------------- 2) phase + resonance log-gate ----------------
__global__ void phase_gate_kernel(const float* __restrict__ x, const float* __restrict__ w_phase,
                                  const float* __restrict__ carrier, const float* __restrict__ lock,
                                  const float* __restrict__ bandwidth, float* __restrict__ loggate) {
  const int wave = threadIdx.x >> 5, lane = threadIdx.x & 31;
  const int token = blockIdx.x * 8 + wave;           // 0 .. B*S-1
  const float* xr = x + (size_t)token * D_;
  float s0 = 0.f, s1 = 0.f;
  for (int i = lane; i < D_; i += 32) {
    float xv = xr[i];
    s0 += xv * w_phase[2 * i + 0];
    s1 += xv * w_phase[2 * i + 1];
  }
  for (int off = 16; off >= 1; off >>= 1) {
    s0 += __shfl_xor(s0, off, 32);
    s1 += __shfl_xor(s1, off, 32);
  }
  float phase = atan2f(s0, s1);
  if (lane < H_) {
    int h = lane;
    float dphi = phase - carrier[h];
    float window = __expf((__cosf(dphi) - 1.0f) / bandwidth[h]);
    float lk = lock[h];
    float gate = lk * window + (1.0f - lk);
    int b = token / S_, s = token % S_;
    loggate[((size_t)b * H_ + h) * S_ + s] = __logf(gate + 1e-6f);
  }
}

// ---------------- 3) tiled WMMA GEMM, double-buffered LDS ----------------
// C[M,N] = A[M,K] * B[K,N]. BM=128, BN=128, BK=32. 8 waves (2Mx4N), wave = 64x32 = 8 wmma.
template <bool F16OUT>
__global__ __launch_bounds__(256, 1) void gemm_wmma(const _Float16* __restrict__ A,
                                                    const _Float16* __restrict__ Bm,
                                                    void* __restrict__ Out, int M, int N, int K) {
  __shared__ _Float16 ldsA[2][128 * 32];    // [buf][row][k]  16 KB
  __shared__ _Float16 ldsBt[2][128 * 32];   // [buf][col][k]  16 KB (transposed)
  const int t = threadIdx.x;
  const int wave = t >> 5, lane = t & 31;
  const int lane16 = lane & 15, half = lane >> 4;
  const int m0 = blockIdx.y * 128, n0 = blockIdx.x * 128;
  const int wm = wave & 1, wn = wave >> 1;   // 2 x 4

  // per-thread staging geometry
  const int arow0 = t >> 2,            acc0 = (t & 3) * 8;          // A chunk 0
  const int arow1 = (t + 256) >> 2,    acc1 = ((t + 256) & 3) * 8;  // A chunk 1
  const int brow0 = t >> 4,            bcc0 = (t & 15) * 8;         // B chunk 0
  const int brow1 = (t + 256) >> 4,    bcc1 = ((t + 256) & 15) * 8; // B chunk 1

  auto loadTile = [&](int kb, u32x4 ra[2], u32x4 rb[2]) {
    ra[0] = *(const u32x4*)(A + (size_t)(m0 + arow0) * K + kb + acc0);
    ra[1] = *(const u32x4*)(A + (size_t)(m0 + arow1) * K + kb + acc1);
    rb[0] = *(const u32x4*)(Bm + (size_t)(kb + brow0) * N + n0 + bcc0);
    rb[1] = *(const u32x4*)(Bm + (size_t)(kb + brow1) * N + n0 + bcc1);
  };
  auto storeTile = [&](int buf, u32x4 ra[2], u32x4 rb[2]) {
    *(u32x4*)(ldsA[buf] + arow0 * 32 + acc0) = ra[0];
    *(u32x4*)(ldsA[buf] + arow1 * 32 + acc1) = ra[1];
    const _Float16* h0 = (const _Float16*)&rb[0];
    const _Float16* h1 = (const _Float16*)&rb[1];
#pragma unroll
    for (int j = 0; j < 8; j++) ldsBt[buf][(bcc0 + j) * 32 + brow0] = h0[j];
#pragma unroll
    for (int j = 0; j < 8; j++) ldsBt[buf][(bcc1 + j) * 32 + brow1] = h1[j];
  };

  v8f acc[4][2];
#pragma unroll
  for (int mt = 0; mt < 4; mt++)
#pragma unroll
    for (int nt = 0; nt < 2; nt++) acc[mt][nt] = (v8f){0, 0, 0, 0, 0, 0, 0, 0};

  const int nk = K >> 5;
  {
    u32x4 ra[2], rb[2];
    loadTile(0, ra, rb);
    storeTile(0, ra, rb);
  }
  __syncthreads();

  for (int kt = 0; kt < nk; ++kt) {
    const int cur = kt & 1;
    u32x4 ra[2], rb[2];
    if (kt + 1 < nk) loadTile((kt + 1) << 5, ra, rb);   // overlap with compute below

    Frag16 af[4], bf[2];
#pragma unroll
    for (int mt = 0; mt < 4; mt++) {
      const _Float16* p = ldsA[cur] + (wm * 64 + mt * 16 + lane16) * 32;
      af[mt].u[0] = *(const u32x4*)(p + half * 8);        // K = half*8 .. +7
      af[mt].u[1] = *(const u32x4*)(p + 16 + half * 8);   // K = 16+half*8 .. +7
    }
#pragma unroll
    for (int nt = 0; nt < 2; nt++) {
      const _Float16* p = ldsBt[cur] + (wn * 32 + nt * 16 + lane16) * 32;
      bf[nt].u[0] = *(const u32x4*)(p + half * 8);        // K = half*8 .. +7
      bf[nt].u[1] = *(const u32x4*)(p + 16 + half * 8);   // K = 16+half*8 .. +7
    }
#pragma unroll
    for (int mt = 0; mt < 4; mt++)
#pragma unroll
      for (int nt = 0; nt < 2; nt++) acc[mt][nt] = wmma_f16(af[mt], bf[nt], acc[mt][nt]);

    if (kt + 1 < nk) storeTile(cur ^ 1, ra, rb);
    __syncthreads();
  }

  // store (C layout: VGPR r -> row r + half*8; lane16 -> col)
#pragma unroll
  for (int mt = 0; mt < 4; mt++)
#pragma unroll
    for (int nt = 0; nt < 2; nt++)
#pragma unroll
      for (int r = 0; r < 8; r++) {
        int row = m0 + wm * 64 + mt * 16 + r + half * 8;
        int col = n0 + wn * 32 + nt * 16 + lane16;
        float vv = acc[mt][nt][r];
        if (F16OUT)
          ((_Float16*)Out)[(size_t)row * N + col] = (_Float16)vv;
        else
          ((float*)Out)[(size_t)row * N + col] = vv;
      }
}

// ---------------- 4) flash causal attention with resonance gate ----------------
// block = (b, h, 128 q rows); 8 waves x 16 q rows each. Double-buffered K/V staging.
__global__ __launch_bounds__(256, 1) void attn_kernel(const _Float16* __restrict__ qh,
                                                      const _Float16* __restrict__ kh,
                                                      const _Float16* __restrict__ vh,
                                                      const float* __restrict__ loggate,
                                                      const int* __restrict__ active_mask,
                                                      _Float16* __restrict__ oh) {
  __shared__ _Float16 ldsK[2][32 * 64];     // [buf][kpos][dh]  8 KB
  __shared__ _Float16 ldsVt[2][64 * 32];    // [buf][dh][kpos]  8 KB (transposed)
  __shared__ _Float16 ldsP[8 * 16 * 32];    // per-wave P tiles, 8 KB
  const int t = threadIdx.x;
  const int wave = t >> 5, lane = t & 31, lane16 = lane & 15, half = lane >> 4;
  const int bid = blockIdx.x;
  const int qblk = bid & 7;
  const int h = (bid >> 3) & (H_ - 1);
  const int b = bid >> 7;
  const int qb = qblk * 128;
  const int q0 = qb + wave * 16;

  // staging geometry: thread covers row kr, 8-half chunk c0 of the 32x64 tile
  const int kr = t >> 3, c0 = (t & 7) * 8;
  const size_t gbase = (size_t)(b * S_) * D_ + h * DH_ + c0;

  auto loadKV = [&](int kbase, u32x4& kv, u32x4& vv) {
    size_t grow = gbase + (size_t)(kbase + kr) * D_;
    kv = *(const u32x4*)(kh + grow);
    vv = *(const u32x4*)(vh + grow);
  };
  auto storeKV = [&](int buf, u32x4 kv, u32x4 vv) {
    *(u32x4*)(ldsK[buf] + kr * 64 + c0) = kv;
    const _Float16* hv = (const _Float16*)&vv;
#pragma unroll
    for (int j = 0; j < 8; j++) ldsVt[buf][(c0 + j) * 32 + kr] = hv[j];
  };

  // Q fragments (A layout, 16x64 = two K=32 frags), straight from global
  Frag16 qa[2];
  {
    const _Float16* qrow = qh + ((size_t)(b * S_ + q0 + lane16)) * D_ + h * DH_;
#pragma unroll
    for (int f = 0; f < 2; f++) {
      qa[f].u[0] = *(const u32x4*)(qrow + f * 32 + half * 8);
      qa[f].u[1] = *(const u32x4*)(qrow + f * 32 + 16 + half * 8);
    }
  }

  float m[8], l[8];
  v8f oacc[4];
#pragma unroll
  for (int r = 0; r < 8; r++) { m[r] = -3.0e38f; l[r] = 0.f; }
#pragma unroll
  for (int n = 0; n < 4; n++) oacc[n] = (v8f){0, 0, 0, 0, 0, 0, 0, 0};

  const int qmax = q0 + 15;
  const int nkt = (qb + 128) >> 5;  // 32-key tiles needed by this block (causal)
  const float scale = 0.125f;       // 1/sqrt(64)
  const float* lg_base = loggate + ((size_t)b * H_ + h) * S_;
  _Float16* pp = ldsP + wave * (16 * 32);

  {
    u32x4 kv, vv;
    loadKV(0, kv, vv);
    storeKV(0, kv, vv);
  }
  __syncthreads();

  for (int kt = 0; kt < nkt; ++kt) {
    const int cur = kt & 1;
    const int kbase = kt * 32;
    u32x4 nkv, nvv;
    if (kt + 1 < nkt) loadKV(kbase + 32, nkv, nvv);   // overlap with compute

    if (kbase <= qmax) {
      // scores: two 16x16 tiles (key cols j*16..), contraction over dh=64 (2 wmma each)
      v8f sc[2];
#pragma unroll
      for (int j = 0; j < 2; j++) {
        Frag16 kb0, kb1;
        const _Float16* kp = ldsK[cur] + (j * 16 + lane16) * 64;   // col = key pos
        kb0.u[0] = *(const u32x4*)(kp + half * 8);
        kb0.u[1] = *(const u32x4*)(kp + 16 + half * 8);
        kb1.u[0] = *(const u32x4*)(kp + 32 + half * 8);
        kb1.u[1] = *(const u32x4*)(kp + 48 + half * 8);
        v8f z = (v8f){0, 0, 0, 0, 0, 0, 0, 0};
        z = wmma_f16(qa[0], kb0, z);
        z = wmma_f16(qa[1], kb1, z);
        sc[j] = z;
      }
      // gate (log domain) + causal mask; mask only on diagonal-straddling tiles
      float lg0 = lg_base[kbase + lane16];
      float lg1 = lg_base[kbase + 16 + lane16];
      if (kbase + 31 <= q0) {          // interior tile: no masking needed
#pragma unroll
        for (int r = 0; r < 8; r++) {
          sc[0][r] = sc[0][r] * scale + lg0;
          sc[1][r] = sc[1][r] * scale + lg1;
        }
      } else {                          // diagonal tile: per-element causal mask
#pragma unroll
        for (int j = 0; j < 2; j++) {
          int kpos = kbase + j * 16 + lane16;
          float lg = j ? lg1 : lg0;
#pragma unroll
          for (int r = 0; r < 8; r++) {
            int qrow = q0 + r + half * 8;
            float v = sc[j][r] * scale + lg;
            sc[j][r] = (kpos <= qrow) ? v : -3.0e38f;
          }
        }
      }
      // online softmax (row = r + half*8, cols across the 16-lane half)
      float p0[8], p1[8];
#pragma unroll
      for (int r = 0; r < 8; r++) {
        float tm = fmaxf(sc[0][r], sc[1][r]);
#pragma unroll
        for (int off = 8; off >= 1; off >>= 1) tm = fmaxf(tm, __shfl_xor(tm, off, 32));
        float mn = fmaxf(m[r], tm);
        float rescale = __expf(m[r] - mn);
        float a0 = __expf(sc[0][r] - mn);
        float a1 = __expf(sc[1][r] - mn);
        float rs = a0 + a1;
#pragma unroll
        for (int off = 8; off >= 1; off >>= 1) rs += __shfl_xor(rs, off, 32);
        l[r] = l[r] * rescale + rs;
        m[r] = mn;
        p0[r] = a0; p1[r] = a1;
#pragma unroll
        for (int n = 0; n < 4; n++) oacc[n][r] *= rescale;
      }
      // P: C layout -> LDS -> A layout
#pragma unroll
      for (int r = 0; r < 8; r++) {
        int row = r + half * 8;
        pp[row * 32 + lane16] = (_Float16)p0[r];
        pp[row * 32 + 16 + lane16] = (_Float16)p1[r];
      }
      asm volatile("s_wait_dscnt 0" ::: "memory");   // in-wave LDS store->load ordering
      Frag16 pa;
      {
        const _Float16* pr = pp + lane16 * 32;
        pa.u[0] = *(const u32x4*)(pr + half * 8);
        pa.u[1] = *(const u32x4*)(pr + 16 + half * 8);
      }
#pragma unroll
      for (int n = 0; n < 4; n++) {
        Frag16 vb;
        const _Float16* vp = ldsVt[cur] + (n * 16 + lane16) * 32;
        vb.u[0] = *(const u32x4*)(vp + half * 8);
        vb.u[1] = *(const u32x4*)(vp + 16 + half * 8);
        oacc[n] = wmma_f16(pa, vb, oacc[n]);
      }
    }

    if (kt + 1 < nkt) storeKV(cur ^ 1, nkv, nvv);
    __syncthreads();
  }

  // epilogue: normalize, apply active_mask, store f16 row-major (col = h*64+dh)
  float am = (active_mask[h] != 0) ? 1.0f : 0.0f;
#pragma unroll
  for (int r = 0; r < 8; r++) {
    float inv = am / l[r];
    int row = q0 + r + half * 8;
    _Float16* orow = oh + ((size_t)(b * S_ + row)) * D_ + h * DH_;
#pragma unroll
    for (int n = 0; n < 4; n++) orow[n * 16 + lane16] = (_Float16)(oacc[n][r] * inv);
  }
}

// ---------------- launcher ----------------
extern "C" void kernel_launch(void* const* d_in, const int* in_sizes, int n_in,
                              void* d_out, int out_size, void* d_ws, size_t ws_size,
                              hipStream_t stream) {
  const float* x = (const float*)d_in[0];
  const float* wq = (const float*)d_in[1];
  const float* wk = (const float*)d_in[2];
  const float* wv = (const float*)d_in[3];
  const float* wo = (const float*)d_in[4];
  const float* w_phase = (const float*)d_in[5];
  const float* carrier = (const float*)d_in[6];
  const float* lock = (const float*)d_in[7];
  const float* bandwidth = (const float*)d_in[8];
  const int* active_mask = (const int*)d_in[9];
  (void)in_sizes; (void)n_in; (void)out_size; (void)ws_size;

  char* ws = (char*)d_ws;
  size_t off = 0;
  auto alloc = [&](size_t bytes) { void* p = ws + off; off += (bytes + 255) & ~(size_t)255; return p; };
  const size_t NX = (size_t)B_ * S_ * D_;   // 4M
  const size_t NW = (size_t)D_ * D_;        // 1M
  _Float16* xh  = (_Float16*)alloc(NX * 2);
  _Float16* wqh = (_Float16*)alloc(NW * 2);
  _Float16* wkh = (_Float16*)alloc(NW * 2);
  _Float16* wvh = (_Float16*)alloc(NW * 2);
  _Float16* woh = (_Float16*)alloc(NW * 2);
  _Float16* qhd = (_Float16*)alloc(NX * 2);
  _Float16* khd = (_Float16*)alloc(NX * 2);
  _Float16* vhd = (_Float16*)alloc(NX * 2);
  _Float16* ohd = (_Float16*)alloc(NX * 2);
  float* loggate = (float*)alloc((size_t)B_ * H_ * S_ * 4);

  // 1) converts
  {
    int n4 = (int)(NX / 4);
    cvt_f32_f16<<<(n4 + 255) / 256, 256, 0, stream>>>((const f32x4*)x, (v4h*)xh, n4);
    int w4 = (int)(NW / 4);
    cvt_f32_f16<<<(w4 + 255) / 256, 256, 0, stream>>>((const f32x4*)wq, (v4h*)wqh, w4);
    cvt_f32_f16<<<(w4 + 255) / 256, 256, 0, stream>>>((const f32x4*)wk, (v4h*)wkh, w4);
    cvt_f32_f16<<<(w4 + 255) / 256, 256, 0, stream>>>((const f32x4*)wv, (v4h*)wvh, w4);
    cvt_f32_f16<<<(w4 + 255) / 256, 256, 0, stream>>>((const f32x4*)wo, (v4h*)woh, w4);
  }
  // 2) phase + gate
  phase_gate_kernel<<<(B_ * S_) / 8, 256, 0, stream>>>(x, w_phase, carrier, lock, bandwidth, loggate);

  // 3) q/k/v projections (M=4096, N=1024, K=1024)
  dim3 ggrid(D_ / 128, (B_ * S_) / 128);
  gemm_wmma<true><<<ggrid, 256, 0, stream>>>(xh, wqh, (void*)qhd, B_ * S_, D_, D_);
  gemm_wmma<true><<<ggrid, 256, 0, stream>>>(xh, wkh, (void*)khd, B_ * S_, D_, D_);
  gemm_wmma<true><<<ggrid, 256, 0, stream>>>(xh, wvh, (void*)vhd, B_ * S_, D_, D_);

  // 4) attention: B*H*(S/128) blocks
  attn_kernel<<<B_ * H_ * (S_ / 128), 256, 0, stream>>>(qhd, khd, vhd, loggate, active_mask, ohd);

  // 5) output projection -> f32 d_out
  gemm_wmma<false><<<ggrid, 256, 0, stream>>>(ohd, woh, d_out, B_ * S_, D_, D_);
}
